// MultiHeadAttention_35485019799766
// MI455X (gfx1250) — matrix operand order repaired and checked by reading, not compiled, not run on hardware
//
#include <hip/hip_runtime.h>
#include <hip/hip_bf16.h>

// ---------------------------------------------------------------------------
// MultiHeadAttention for MI455X (gfx1250, wave32, WMMA f32_16x16x32_f16)
//   B=2, S=2048, D=1024, H=16, d_k=64, causal mask, softmax, out-proj.
// Pipeline:
//   (0) f32 -> f16 conversion of activations + weights (once)
//   (1) Q/K/V projections: WMMA GEMM, B stripe staged to LDS with
//       GLOBAL_LOAD_ASYNC_TO_LDS_B128 (ASYNCcnt, double-buffered),
//       32(M) x 64(N) wave tile -> 8 back-to-back WMMAs per K-step
//   (2) flash-attention per (b,h, 16-query tile), P staged via LDS
//   (3) output projection: WMMA GEMM, fp32 result
// ---------------------------------------------------------------------------

typedef _Float16 half_t;
typedef __attribute__((ext_vector_type(16))) _Float16 v16h;
typedef __attribute__((ext_vector_type(8)))  _Float16 v8h;
typedef __attribute__((ext_vector_type(4)))  _Float16 v4h;
typedef __attribute__((ext_vector_type(8)))  float    v8f;
typedef __attribute__((ext_vector_type(4)))  float    v4f;

#define DMODEL 1024
#define SLEN   2048
#define NHEAD  16
#define DK     64
#define MROWS  4096           // B * S

__device__ __forceinline__ v8f wmma_f16(v16h a, v16h b, v8f c) {
    // emits v_wmma_f32_16x16x32_f16
    return __builtin_amdgcn_wmma_f32_16x16x32_f16(
        /*neg_a=*/false, a, /*neg_b=*/false, b,
        /*c_mod=*/(short)0, c, /*reuse_a=*/false, /*reuse_b=*/false);
}

// Async copy 16B/lane from global to LDS, tracked by ASYNCcnt.
// (CDNA5 ISA: GLOBAL_LOAD_ASYNC_TO_LDS_B128, GV mode)
__device__ __forceinline__ void async_g2l_b128(unsigned lds_off, const void* gaddr) {
    asm volatile("global_load_async_to_lds_b128 %0, %1, off"
                 :: "v"(lds_off), "v"(gaddr) : "memory");
}
__device__ __forceinline__ void wait_async0() {
    asm volatile("s_wait_asynccnt 0x0" ::: "memory");
}
__device__ __forceinline__ void wait_async1() {
    asm volatile("s_wait_asynccnt 0x1" ::: "memory");
}

// A fragment, 16x32 f16 (M x K). Lane L holds row M = L&15 with 16
// contiguous K halves starting at kbase + (L>=16 ? 16 : 0).
__device__ __forceinline__ v16h load_frag_a(const half_t* __restrict__ base,
                                            int ld, int row0, int kbase, int lane) {
    const int r    = row0 + (lane & 15);
    const int koff = kbase + ((lane >> 4) << 4);
    const half_t* p = base + (size_t)r * ld + koff;
    return *(const v16h*)(p);
}

// B fragment, 32x16 f16 (K x N). Lane L holds column N = L&15 with 16
// contiguous K halves starting at kbase + (L>=16 ? 16 : 0).
// base is row-major with "rows = N" (i.e. stores B^T), ld = its row stride.
__device__ __forceinline__ v16h load_frag_b(const half_t* __restrict__ base,
                                            int ld, int col0, int kbase, int lane) {
    const int c    = col0 + (lane & 15);
    const int koff = kbase + ((lane >> 4) << 4);
    const half_t* p = base + (size_t)c * ld + koff;
    return *(const v16h*)(p);   // 32B: two b128 loads
}

// B fragment from an LDS-resident B^T tile (64 rows x 32 halves, ld = 32).
__device__ __forceinline__ v16h load_frag_b_lds(const _Float16* base,
                                                int col0, int lane) {
    const int c    = col0 + (lane & 15);
    const int koff = (lane >> 4) << 4;
    return *(const v16h*)(base + c * 32 + koff);
}

// A fragment from a 16x32 f16 LDS tile (row-major, ld = 32 halves).
__device__ __forceinline__ v16h load_frag_a_lds(const _Float16* P, int lane) {
    const int r    = lane & 15;
    const int koff = (lane >> 4) << 4;
    const _Float16* p = P + r * 32 + koff;
    return *(const v16h*)(p);
}

// ---------------------------------------------------------------------------
// (0) fp32 -> fp16 conversion, 4 elements / thread
// ---------------------------------------------------------------------------
__global__ __launch_bounds__(256) void k_f32_to_f16(const float* __restrict__ in,
                                                    half_t* __restrict__ out, int n4) {
    int i = blockIdx.x * 256 + threadIdx.x;
    if (i < n4) {
        v4f x = *(const v4f*)(in + 4 * (size_t)i);
        v4h y;
#pragma unroll
        for (int j = 0; j < 4; ++j) y[j] = (_Float16)x[j];
        *(v4h*)(out + 4 * (size_t)i) = y;
    }
}

// ---------------------------------------------------------------------------
// (1)/(3) GEMM:  Y[m,e] = sum_k X[m,k] * W[e,k] + bias[e]
//   X: [4096,1024] f16 row-major, W: [1024,1024] f16 row-major (nn.Linear W)
//   mode 0: f16 out, layout [B,H,S,DK]        (Q, K)
//   mode 1: f16 out, layout [B,H,DK,S]        (V transposed for PV B-frags)
//   mode 2: f32 out, layout [4096,1024]       (final projection -> d_out)
// Block = 256 threads (8 waves); wave tile 32(M) x 64(N); block 256 x 64.
// The shared 64x32 B stripe (4 KB) is staged once per block into a 2-deep
// LDS ring via async-to-LDS DMA, overlapped with WMMA on the previous stripe.
// Per K-step: preload 2 A-frags + 4 B-frags, then 8 back-to-back WMMAs.
// ---------------------------------------------------------------------------
__global__ __launch_bounds__(256) void k_gemm_xwt(const half_t* __restrict__ X,
                                                  const half_t* __restrict__ W,
                                                  const float* __restrict__ bias,
                                                  half_t* __restrict__ outh,
                                                  float* __restrict__ outf,
                                                  int mode) {
    const int tid  = threadIdx.x;
    const int lane = tid & 31;
    const int wave = tid >> 5;
    const int r0 = blockIdx.y * 256 + wave * 32;
    const int c0 = blockIdx.x * 64;

    __shared__ __align__(16) _Float16 Bt[2][64 * 32];   // 2 x 4 KB ring

    // each thread DMAs one b128: col = tid>>2 (0..63), k-chunk = tid&3
    const int scol = tid >> 2;
    const int sch  = (tid & 3) * 8;
    const half_t* wsrc = W + (size_t)(c0 + scol) * DMODEL + sch;
    const unsigned ldsA = (unsigned)(uintptr_t)(&Bt[0][0] + scol * 32 + sch);
    const unsigned ldsB = (unsigned)(uintptr_t)(&Bt[1][0] + scol * 32 + sch);

    v8f acc[2][4];
#pragma unroll
    for (int rh = 0; rh < 2; ++rh)
#pragma unroll
        for (int t = 0; t < 4; ++t) acc[rh][t] = (v8f)(0.0f);

    async_g2l_b128(ldsA, wsrc);                       // stage 0
    const int NK = DMODEL / 32;                       // 32 K-steps
    for (int kt = 0; kt < NK; ++kt) {
        const int k = kt * 32;
        if (kt + 1 < NK) {                            // stage kt+1, keep it in flight
            async_g2l_b128((kt & 1) ? ldsA : ldsB, wsrc + (k + 32));
            wait_async1();                            // stripe kt has landed
        } else {
            wait_async0();
        }
        __syncthreads();

        // preload all fragments, then run the 8-WMMA chain uninterrupted
        v16h a0 = load_frag_a(X, DMODEL, r0,      k, lane);
        v16h a1 = load_frag_a(X, DMODEL, r0 + 16, k, lane);
        const _Float16* bt = &Bt[kt & 1][0];
        v16h bfr[4];
#pragma unroll
        for (int t = 0; t < 4; ++t) bfr[t] = load_frag_b_lds(bt, 16 * t, lane);
#pragma unroll
        for (int t = 0; t < 4; ++t) {
            acc[0][t] = wmma_f16(a0, bfr[t], acc[0][t]);
            acc[1][t] = wmma_f16(a1, bfr[t], acc[1][t]);
        }
        __syncthreads();                              // reads done before ring reuse
    }

    const int n   = lane & 15;
    const int hi8 = (lane >> 4) << 3;
#pragma unroll
    for (int rh = 0; rh < 2; ++rh) {
#pragma unroll
        for (int t = 0; t < 4; ++t) {
#pragma unroll
            for (int i = 0; i < 8; ++i) {
                const int mg = r0 + rh * 16 + hi8 + i;
                const int e  = c0 + 16 * t + n;
                float v = acc[rh][t][i] + bias[e];
                if (mode == 2) {
                    outf[(size_t)mg * DMODEL + e] = v;
                } else {
                    const int b = mg >> 11, s = mg & (SLEN - 1);
                    const int h = e >> 6,  d = e & (DK - 1);
                    if (mode == 0)
                        outh[(((size_t)(b * NHEAD + h)) * SLEN + s) * DK + d] = (_Float16)v;
                    else
                        outh[(((size_t)(b * NHEAD + h)) * DK + d) * SLEN + s] = (_Float16)v;
                }
            }
        }
    }
}

// ---------------------------------------------------------------------------
// (2) causal flash attention. One wave per (b,h, 16-query tile).
//   q16/k16: [B,H,S,DK] f16.  v16t: [B,H,DK,S] f16 (transposed).
//   out16:   [4096, 1024] f16 (heads re-concatenated) for the final GEMM.
// ---------------------------------------------------------------------------
__global__ __launch_bounds__(32) void k_attention(const half_t* __restrict__ q16,
                                                  const half_t* __restrict__ k16,
                                                  const half_t* __restrict__ v16t,
                                                  half_t* __restrict__ out16) {
    const int lane = threadIdx.x & 31;
    const int qt   = blockIdx.x;          // 0..127
    const int bh   = blockIdx.y;          // 0..31
    const int q0   = qt * 16;
    const int n    = lane & 15;
    const int hi8  = (lane >> 4) << 3;

    const half_t* qb = q16  + (size_t)bh * SLEN * DK;
    const half_t* kb = k16  + (size_t)bh * SLEN * DK;
    const half_t* vb = v16t + (size_t)bh * DK * SLEN;

    __shared__ _Float16 Pt[16 * 32];      // one 16x32 P tile (single wave/block)

    const v16h qa0 = load_frag_a(qb, DK, q0,  0, lane);
    const v16h qa1 = load_frag_a(qb, DK, q0, 32, lane);

    v8f o[4];
#pragma unroll
    for (int t = 0; t < 4; ++t) o[t] = (v8f)(0.0f);
    float mrow[8], lrow[8];
#pragma unroll
    for (int i = 0; i < 8; ++i) { mrow[i] = -1e30f; lrow[i] = 0.0f; }

    const int ktmax = (q0 + 15) >> 5;     // causal: keys <= q0+15
    for (int kt = 0; kt <= ktmax; ++kt) {
        const int k0 = kt * 32;

        // ---- scores S = Q K^T / sqrt(dk), two 16x16 C tiles (keys k0.., k0+16..)
        // preload all 4 K fragments, then 4 back-to-back WMMAs
        v16h kf[4];
        kf[0] = load_frag_b(kb, DK, k0,      0,  lane);
        kf[1] = load_frag_b(kb, DK, k0,      32, lane);
        kf[2] = load_frag_b(kb, DK, k0 + 16, 0,  lane);
        kf[3] = load_frag_b(kb, DK, k0 + 16, 32, lane);
        v8f s0 = (v8f)(0.0f), s1 = (v8f)(0.0f);
        s0 = wmma_f16(qa0, kf[0], s0);
        s0 = wmma_f16(qa1, kf[1], s0);
        s1 = wmma_f16(qa0, kf[2], s1);
        s1 = wmma_f16(qa1, kf[3], s1);

        // ---- scale + causal mask, per-row max
        float rmax[8];
#pragma unroll
        for (int i = 0; i < 8; ++i) {
            const int qg = q0 + hi8 + i;
            float a = s0[i] * 0.125f;  if (k0 + n      > qg) a = -1e9f;
            float b = s1[i] * 0.125f;  if (k0 + 16 + n > qg) b = -1e9f;
            s0[i] = a; s1[i] = b;
            rmax[i] = fmaxf(a, b);
        }
#pragma unroll
        for (int off = 1; off < 16; off <<= 1)
#pragma unroll
            for (int i = 0; i < 8; ++i)
                rmax[i] = fmaxf(rmax[i], __shfl_xor(rmax[i], off, 32));

        // ---- online softmax update
        float alpha[8], rsum[8];
#pragma unroll
        for (int i = 0; i < 8; ++i) {
            const float mn = fmaxf(mrow[i], rmax[i]);
            alpha[i] = __expf(mrow[i] - mn);
            mrow[i]  = mn;
            const float p0 = __expf(s0[i] - mn);
            const float p1 = __expf(s1[i] - mn);
            s0[i] = p0; s1[i] = p1;
            rsum[i] = p0 + p1;
        }
#pragma unroll
        for (int off = 1; off < 16; off <<= 1)
#pragma unroll
            for (int i = 0; i < 8; ++i)
                rsum[i] += __shfl_xor(rsum[i], off, 32);
#pragma unroll
        for (int i = 0; i < 8; ++i) lrow[i] = lrow[i] * alpha[i] + rsum[i];
#pragma unroll
        for (int t = 0; t < 4; ++t)
#pragma unroll
            for (int i = 0; i < 8; ++i) o[t][i] *= alpha[i];

        // ---- stage P (C layout -> row-major f16 tile) through LDS
        __syncthreads();
#pragma unroll
        for (int i = 0; i < 8; ++i) {
            Pt[(hi8 + i) * 32 + n]      = (_Float16)s0[i];
            Pt[(hi8 + i) * 32 + 16 + n] = (_Float16)s1[i];
        }
        __syncthreads();
        const v16h pa = load_frag_a_lds(Pt, lane);

        // ---- O += P V   (B-frags from transposed V: contiguous over keys)
        v16h vf[4];
#pragma unroll
        for (int t = 0; t < 4; ++t) vf[t] = load_frag_b(vb, SLEN, 16 * t, k0, lane);
#pragma unroll
        for (int t = 0; t < 4; ++t) o[t] = wmma_f16(pa, vf[t], o[t]);
    }

    // ---- normalize and store, re-concatenating heads: out16[b*S+s, h*64+d]
    const int b = bh >> 4, h = bh & 15;
#pragma unroll
    for (int t = 0; t < 4; ++t) {
#pragma unroll
        for (int i = 0; i < 8; ++i) {
            const int s = q0 + hi8 + i;
            const float v = o[t][i] / lrow[i];
            out16[((size_t)(b * SLEN + s)) * DMODEL + h * DK + 16 * t + n] = (_Float16)v;
        }
    }
}

// ---------------------------------------------------------------------------
// launch
// ---------------------------------------------------------------------------
extern "C" void kernel_launch(void* const* d_in, const int* in_sizes, int n_in,
                              void* d_out, int out_size, void* d_ws, size_t ws_size,
                              hipStream_t stream) {
    const float* Q   = (const float*)d_in[0];
    const float* W_Q = (const float*)d_in[1];
    const float* b_Q = (const float*)d_in[2];
    const float* W_K = (const float*)d_in[3];
    const float* b_K = (const float*)d_in[4];
    const float* W_V = (const float*)d_in[5];
    const float* b_V = (const float*)d_in[6];
    const float* W_O = (const float*)d_in[7];
    const float* b_O = (const float*)d_in[8];
    // d_in[9] = mask: causal, recomputed analytically in-kernel.

    char* ws = (char*)d_ws;
    size_t off = 0;
    auto take = [&](size_t bytes) { char* p = ws + off; off += bytes; return p; };
    const size_t ACT = (size_t)MROWS * DMODEL * sizeof(half_t);   // 8 MiB
    const size_t WGT = (size_t)DMODEL * DMODEL * sizeof(half_t);  // 2 MiB

    half_t* X16   = (half_t*)take(ACT);
    half_t* Wq16  = (half_t*)take(WGT);
    half_t* Wk16  = (half_t*)take(WGT);
    half_t* Wv16  = (half_t*)take(WGT);
    half_t* Wo16  = (half_t*)take(WGT);
    half_t* q16   = (half_t*)take(ACT);
    half_t* k16   = (half_t*)take(ACT);
    half_t* v16t  = (half_t*)take(ACT);
    half_t* att16 = (half_t*)take(ACT);
    (void)ws_size; (void)n_in; (void)in_sizes; (void)out_size;

    const int nAct4 = MROWS * DMODEL / 4;     // 1,048,576
    const int nWgt4 = DMODEL * DMODEL / 4;    //   262,144
    k_f32_to_f16<<<nAct4 / 256, 256, 0, stream>>>(Q,   X16,  nAct4);
    k_f32_to_f16<<<nWgt4 / 256, 256, 0, stream>>>(W_Q, Wq16, nWgt4);
    k_f32_to_f16<<<nWgt4 / 256, 256, 0, stream>>>(W_K, Wk16, nWgt4);
    k_f32_to_f16<<<nWgt4 / 256, 256, 0, stream>>>(W_V, Wv16, nWgt4);
    k_f32_to_f16<<<nWgt4 / 256, 256, 0, stream>>>(W_O, Wo16, nWgt4);

    dim3 ggrid(DMODEL / 64, MROWS / 256);     // 16 x 16 blocks, 8 waves each
    k_gemm_xwt<<<ggrid, 256, 0, stream>>>(X16, Wq16, b_Q, q16,  nullptr, 0);
    k_gemm_xwt<<<ggrid, 256, 0, stream>>>(X16, Wk16, b_K, k16,  nullptr, 0);
    k_gemm_xwt<<<ggrid, 256, 0, stream>>>(X16, Wv16, b_V, v16t, nullptr, 1);

    dim3 agrid(SLEN / 16, 2 * NHEAD);         // 128 query tiles x 32 (b,h)
    k_attention<<<agrid, 32, 0, stream>>>(q16, k16, v16t, att16);

    k_gemm_xwt<<<ggrid, 256, 0, stream>>>(att16, Wo16, b_O, nullptr, (float*)d_out, 2);
}
